// GNpool_60730837565913
// MI455X (gfx1250) — matrix operand order repaired and checked by reading, not compiled, not run on hardware
//
#include <hip/hip_runtime.h>
#include <hip/hip_bf16.h>
#include <stdint.h>

// ---------------- problem constants ----------------
#define N_NODES  50000
#define N_EDGES  800000
#define N_NF     128
#define N_EF     64
#define MSG_DIM  128
#define HID      300
#define HIDP     320          // HIDDEN padded to multiple of 32 (WMMA K-step)
#define EDGE_IN  320          // 2*N_NF + N_EF
#define NODE_IN  256          // N_NF + MSG_DIM
#define NH       128
#define NPRED    16
#define NG       32

#define TILE_M   128          // rows per workgroup tile (8 waves x 16)
#define BLK      256          // 8 wave32 waves

typedef __attribute__((ext_vector_type(16))) __bf16 v16bf;
typedef __attribute__((ext_vector_type(8)))  float  v8f;

// round-to-nearest-even f32 -> bf16 (raw u16)
__device__ inline unsigned short f2bf(float f) {
  unsigned u = __float_as_uint(f);
  u += 0x7FFFu + ((u >> 16) & 1u);
  return (unsigned short)(u >> 16);
}
__device__ inline unsigned pack2bf(float a, float b) {
  return (unsigned)f2bf(a) | ((unsigned)f2bf(b) << 16);
}
// store 4 floats as 4 packed bf16 (8 bytes) -> ds_store_b64
__device__ inline void store_f4_bf16(unsigned short* dst, float4 v) {
  uint2 p;
  p.x = pack2bf(v.x, v.y);
  p.y = pack2bf(v.z, v.w);
  *(uint2*)dst = p;
}

union FragU { uint4 q[2]; v16bf v; };

// A-matrix 16x32 bf16 (ISA 7.12.2): lanes 0-15 -> M=lane, chunks K[0:8],K[16:24];
// lanes 16-31 -> M=lane-16, chunks K[8:16],K[24:32].  base = row*ld + k0
__device__ inline v16bf load_a_frag(const unsigned short* base, int hb) {
  const unsigned short* p = base + hb * 8;
  FragU f;
  f.q[0] = *(const uint4*)(p);
  f.q[1] = *(const uint4*)(p + 16);
  return f.v;
}
// B-matrix 32x16 bf16: lane = column N; lanes 0-15 hold K[0:8],K[16:24],
// lanes 16-31 hold K[8:16],K[24:32].  Wt stored as [N][Kpad] (K contiguous).
__device__ inline v16bf load_b_frag(const unsigned short* base, int hb) {
  const unsigned short* p = base + hb * 8;
  FragU f;
  f.q[0] = *(const uint4*)(p);
  f.q[1] = *(const uint4*)(p + 16);
  return f.v;
}

// One dense layer on a 128-row LDS tile: Aout[128][Nout] = act(Ain[128][K] @ Wt^T + bias)
__device__ inline void mlp_layer_lds(const unsigned short* __restrict__ Wt,   // [Nout][K] bf16 (global, L2-resident)
                                     const float* __restrict__ bias,          // [Nout]
                                     const unsigned short* Ain,               // LDS [128][K]
                                     unsigned short* Aout,                    // LDS [128][Nout]
                                     int K, int Nout, bool relu) {
  const int lane = threadIdx.x & 31;
  const int wave = threadIdx.x >> 5;
  const int r    = lane & 15;
  const int hb   = lane >> 4;
  const int m0   = wave * 16;
  const unsigned short* arow = Ain + (m0 + r) * K;
  for (int n0 = 0; n0 < Nout; n0 += 16) {
    const unsigned short* brow = Wt + (n0 + r) * K;
    if (n0 + 16 < Nout) __builtin_prefetch(Wt + (n0 + 16 + r) * K, 0, 1);   // global_prefetch_b8
    v8f acc = {};
    for (int k0 = 0; k0 < K; k0 += 32) {
      v16bf a = load_a_frag(arow + k0, hb);
      v16bf b = load_b_frag(brow + k0, hb);
      acc = __builtin_amdgcn_wmma_f32_16x16x32_bf16(false, a, false, b, (short)0, acc, false, false);
    }
    const float bv = bias[n0 + r];
    #pragma unroll
    for (int rr = 0; rr < 8; ++rr) {           // D: M = rr + 8*hb, N = n0 + r
      const int m = m0 + rr + hb * 8;
      float v = acc[rr] + bv;
      if (relu) v = fmaxf(v, 0.f);
      Aout[m * Nout + (n0 + r)] = f2bf(v);
    }
  }
}

// ---------------- prep: transpose+pad weights to bf16 Wt[Np][Kp], pad bias ----------------
__global__ void k_prep_w(const float* __restrict__ W, const float* __restrict__ b,
                         unsigned short* __restrict__ Wt, float* __restrict__ bp,
                         int K, int N, int Kp, int Np) {
  const int total = Np * Kp;
  for (int idx = blockIdx.x * blockDim.x + threadIdx.x; idx < total;
       idx += gridDim.x * blockDim.x) {
    const int n = idx / Kp, k = idx - n * Kp;
    const float v = (n < N && k < K) ? W[(size_t)k * N + n] : 0.f;
    Wt[idx] = f2bf(v);
    if (idx < Np) bp[idx] = (idx < N) ? b[idx] : 0.f;
  }
}

__global__ void k_zero4(float4* __restrict__ p, int n4) {   // bulk zero, b128 stores
  for (int i = blockIdx.x * blockDim.x + threadIdx.x; i < n4; i += gridDim.x * blockDim.x)
    p[i] = make_float4(0.f, 0.f, 0.f, 0.f);
}
__global__ void k_zero(float* __restrict__ p, int n) {
  for (int i = blockIdx.x * blockDim.x + threadIdx.x; i < n; i += gridDim.x * blockDim.x)
    p[i] = 0.f;
}

__global__ void k_counts(const int* __restrict__ batch, float* __restrict__ counts) {
  int i = blockIdx.x * blockDim.x + threadIdx.x;
  if (i < N_NODES) atomicAdd(&counts[batch[i]], 1.0f);
}

// ---------------- edge MLP, fused gather + 4 layers + scatter-add ----------------
__global__ __launch_bounds__(BLK) void k_edge_mlp(
    const float* __restrict__ x, const int* __restrict__ ei, const float* __restrict__ ea,
    const unsigned short* __restrict__ Wt1, const float* __restrict__ b1,
    const unsigned short* __restrict__ Wt2, const float* __restrict__ b2,
    const unsigned short* __restrict__ Wt3, const float* __restrict__ b3,
    const unsigned short* __restrict__ Wt4, const float* __restrict__ b4,
    float* __restrict__ aggr) {
  extern __shared__ unsigned short smem[];
  unsigned short* buf0 = smem;                    // 128 x 320 bf16
  unsigned short* buf1 = smem + TILE_M * HIDP;    // 128 x 320 bf16
  const int e0 = blockIdx.x * TILE_M;             // 800000 / 128 = 6250 exact tiles
  const int* srcIdx = ei;                          // edge_index[0]
  const int* dstIdx = ei + N_EDGES;                // edge_index[1]

  // gather msg_in = [x[dst] | x[src] | edge_attr] as float4 -> packed bf16 LDS tile.
  // 2 threads per row; each owns a contiguous 40-float4 half-row; edge indices loaded once.
  {
    const int row  = threadIdx.x >> 1;            // 0..127
    const int half = threadIdx.x & 1;
    const int e    = e0 + row;
    const int dn   = dstIdx[e];
    const int sn   = srcIdx[e];
    const float4* xd  = (const float4*)(x) + (size_t)dn * (N_NF / 4);
    const float4* xs  = (const float4*)(x) + (size_t)sn * (N_NF / 4);
    const float4* eav = (const float4*)(ea) + (size_t)e * (N_EF / 4);
    unsigned short* orow = buf0 + row * EDGE_IN;
    #pragma unroll
    for (int j = 0; j < EDGE_IN / 8; ++j) {       // 40 float4 per half-row
      const int q = half * (EDGE_IN / 8) + j;     // 0..79
      float4 v;
      if (q < 32)       v = xd[q];
      else if (q < 64)  v = xs[q - 32];
      else              v = eav[q - 64];
      store_f4_bf16(orow + q * 4, v);
    }
  }
  __syncthreads();
  mlp_layer_lds(Wt1, b1, buf0, buf1, EDGE_IN, HIDP, true);  __syncthreads();
  mlp_layer_lds(Wt2, b2, buf1, buf0, HIDP,    HIDP, true);  __syncthreads();
  mlp_layer_lds(Wt3, b3, buf0, buf1, HIDP,    HIDP, true);  __syncthreads();

  // layer 4 (no ReLU) fused with segment-sum: atomic f32 add into aggr[dst]
  const int lane = threadIdx.x & 31, wave = threadIdx.x >> 5;
  const int r = lane & 15, hb = lane >> 4, m0 = wave * 16;
  const unsigned short* arow = buf1 + (m0 + r) * HIDP;
  for (int n0 = 0; n0 < MSG_DIM; n0 += 16) {
    const unsigned short* brow = Wt4 + (n0 + r) * HIDP;
    v8f acc = {};
    for (int k0 = 0; k0 < HIDP; k0 += 32) {
      v16bf a = load_a_frag(arow + k0, hb);
      v16bf b = load_b_frag(brow + k0, hb);
      acc = __builtin_amdgcn_wmma_f32_16x16x32_bf16(false, a, false, b, (short)0, acc, false, false);
    }
    const float bv = b4[n0 + r];
    #pragma unroll
    for (int rr = 0; rr < 8; ++rr) {
      const int m = m0 + rr + hb * 8;
      const int node = dstIdx[e0 + m];
      atomicAdd(&aggr[(size_t)node * MSG_DIM + (n0 + r)], acc[rr] + bv);
    }
  }
}

// ---------------- node MLP, fused gather + 4 layers + per-graph pooling sum ----------------
__global__ __launch_bounds__(BLK) void k_node_mlp(
    const float* __restrict__ x, const float* __restrict__ aggr, const int* __restrict__ batch,
    const unsigned short* __restrict__ Wt1, const float* __restrict__ b1,
    const unsigned short* __restrict__ Wt2, const float* __restrict__ b2,
    const unsigned short* __restrict__ Wt3, const float* __restrict__ b3,
    const unsigned short* __restrict__ Wt4, const float* __restrict__ b4,
    float* __restrict__ sums) {
  extern __shared__ unsigned short smem[];
  unsigned short* buf0 = smem;
  unsigned short* buf1 = smem + TILE_M * HIDP;
  const int v0 = blockIdx.x * TILE_M;

  // gather [x[node] | aggr[node]] as float4 -> packed bf16; 2 threads per row.
  {
    const int row  = threadIdx.x >> 1;
    const int half = threadIdx.x & 1;
    const int node = v0 + row;
    const bool ok  = node < N_NODES;
    const float4* xr = (const float4*)(x)    + (size_t)(ok ? node : 0) * (N_NF / 4);
    const float4* ar = (const float4*)(aggr) + (size_t)(ok ? node : 0) * (MSG_DIM / 4);
    unsigned short* orow = buf0 + row * NODE_IN;
    #pragma unroll
    for (int j = 0; j < NODE_IN / 8; ++j) {       // 32 float4 per half-row
      const int q = half * (NODE_IN / 8) + j;     // 0..63
      float4 v = make_float4(0.f, 0.f, 0.f, 0.f);
      if (ok) v = (q < 32) ? xr[q] : ar[q - 32];
      store_f4_bf16(orow + q * 4, v);
    }
  }
  __syncthreads();
  mlp_layer_lds(Wt1, b1, buf0, buf1, NODE_IN, HIDP, true);  __syncthreads();
  mlp_layer_lds(Wt2, b2, buf1, buf0, HIDP,    HIDP, true);  __syncthreads();
  mlp_layer_lds(Wt3, b3, buf0, buf1, HIDP,    HIDP, true);  __syncthreads();

  const int lane = threadIdx.x & 31, wave = threadIdx.x >> 5;
  const int r = lane & 15, hb = lane >> 4, m0 = wave * 16;
  const unsigned short* arow = buf1 + (m0 + r) * HIDP;
  for (int n0 = 0; n0 < NH; n0 += 16) {
    const unsigned short* brow = Wt4 + (n0 + r) * HIDP;
    v8f acc = {};
    for (int k0 = 0; k0 < HIDP; k0 += 32) {
      v16bf a = load_a_frag(arow + k0, hb);
      v16bf b = load_b_frag(brow + k0, hb);
      acc = __builtin_amdgcn_wmma_f32_16x16x32_bf16(false, a, false, b, (short)0, acc, false, false);
    }
    const float bv = b4[n0 + r];
    #pragma unroll
    for (int rr = 0; rr < 8; ++rr) {
      const int m = m0 + rr + hb * 8;
      const int node = v0 + m;
      if (node < N_NODES) {
        const int g = batch[node];
        atomicAdd(&sums[(size_t)g * NH + (n0 + r)], acc[rr] + bv);
      }
    }
  }
}

// ---------------- head: pooled = sums/counts; out = pooled @ linW + linb ----------------
__global__ void k_final(const float* __restrict__ sums, const float* __restrict__ counts,
                        const float* __restrict__ linW, const float* __restrict__ linb,
                        float* __restrict__ out) {
  const int idx = blockIdx.x * blockDim.x + threadIdx.x;
  if (idx >= NG * NPRED) return;
  const int g = idx / NPRED, o = idx - g * NPRED;
  const float inv = 1.0f / fmaxf(counts[g], 1.0f);
  float acc = linb[o];
  for (int c = 0; c < NH; ++c)
    acc = fmaf(sums[(size_t)g * NH + c] * inv, linW[(size_t)c * NPRED + o], acc);
  out[idx] = acc;
}

// ---------------- host orchestration ----------------
extern "C" void kernel_launch(void* const* d_in, const int* in_sizes, int n_in,
                              void* d_out, int out_size, void* d_ws, size_t ws_size,
                              hipStream_t stream) {
  (void)in_sizes; (void)n_in; (void)out_size; (void)ws_size;
  const float* x     = (const float*)d_in[0];
  const int*   ei    = (const int*)d_in[1];
  const float* ea    = (const float*)d_in[2];
  const int*   batch = (const int*)d_in[3];
  const float* mW[4] = {(const float*)d_in[4], (const float*)d_in[6], (const float*)d_in[8],  (const float*)d_in[10]};
  const float* mb[4] = {(const float*)d_in[5], (const float*)d_in[7], (const float*)d_in[9],  (const float*)d_in[11]};
  const float* nW[4] = {(const float*)d_in[12], (const float*)d_in[14], (const float*)d_in[16], (const float*)d_in[18]};
  const float* nb[4] = {(const float*)d_in[13], (const float*)d_in[15], (const float*)d_in[17], (const float*)d_in[19]};
  const float* linW  = (const float*)d_in[20];
  const float* linb  = (const float*)d_in[21];
  float* out = (float*)d_out;

  // workspace carve-up (256B aligned)
  char* ws = (char*)d_ws;
  size_t off = 0;
  auto take = [&](size_t bytes) { char* p = ws + off; off = (off + bytes + 255) & ~(size_t)255; return p; };
  unsigned short* mWt[4]; float* mbp[4];
  unsigned short* nWt[4]; float* nbp[4];
  const int mK[4] = {EDGE_IN, HID, HID, HID}, mN[4] = {HID, HID, HID, MSG_DIM};
  const int mKp[4] = {EDGE_IN, HIDP, HIDP, HIDP}, mNp[4] = {HIDP, HIDP, HIDP, MSG_DIM};
  const int nK[4] = {NODE_IN, HID, HID, HID}, nN[4] = {HID, HID, HID, NH};
  const int nKp[4] = {NODE_IN, HIDP, HIDP, HIDP}, nNp[4] = {HIDP, HIDP, HIDP, NH};
  for (int l = 0; l < 4; ++l) { mWt[l] = (unsigned short*)take((size_t)mNp[l] * mKp[l] * 2); mbp[l] = (float*)take((size_t)mNp[l] * 4); }
  for (int l = 0; l < 4; ++l) { nWt[l] = (unsigned short*)take((size_t)nNp[l] * nKp[l] * 2); nbp[l] = (float*)take((size_t)nNp[l] * 4); }
  float* aggr   = (float*)take((size_t)N_NODES * MSG_DIM * 4);
  float* sums   = (float*)take((size_t)NG * NH * 4);
  float* counts = (float*)take((size_t)NG * 4);

  // 1) zero accumulators (every call; atomics accumulate)
  k_zero4<<<dim3((N_NODES * MSG_DIM / 4 + 255) / 256), dim3(256), 0, stream>>>(
      (float4*)aggr, N_NODES * MSG_DIM / 4);
  k_zero<<<dim3((NG * NH + 255) / 256), dim3(256), 0, stream>>>(sums, NG * NH);
  k_zero<<<dim3(1), dim3(64), 0, stream>>>(counts, NG);

  // 2) transpose/pad weights to bf16 Wt[N][Kpad]
  for (int l = 0; l < 4; ++l)
    k_prep_w<<<dim3((mNp[l] * mKp[l] + 255) / 256), dim3(256), 0, stream>>>(
        mW[l], mb[l], mWt[l], mbp[l], mK[l], mN[l], mKp[l], mNp[l]);
  for (int l = 0; l < 4; ++l)
    k_prep_w<<<dim3((nNp[l] * nKp[l] + 255) / 256), dim3(256), 0, stream>>>(
        nW[l], nb[l], nWt[l], nbp[l], nK[l], nN[l], nKp[l], nNp[l]);

  // 3) per-graph node counts
  k_counts<<<dim3((N_NODES + 255) / 256), dim3(256), 0, stream>>>(batch, counts);

  // 4) edge MLP + scatter-add (WMMA bf16)
  const size_t smemBytes = (size_t)2 * TILE_M * HIDP * sizeof(unsigned short);  // 160 KB of 320 KB WGP LDS
  k_edge_mlp<<<dim3(N_EDGES / TILE_M), dim3(BLK), smemBytes, stream>>>(
      x, ei, ea, mWt[0], mbp[0], mWt[1], mbp[1], mWt[2], mbp[2], mWt[3], mbp[3], aggr);

  // 5) node MLP + pooled sums (WMMA bf16)
  k_node_mlp<<<dim3((N_NODES + TILE_M - 1) / TILE_M), dim3(BLK), smemBytes, stream>>>(
      x, aggr, batch, nWt[0], nbp[0], nWt[1], nbp[1], nWt[2], nbp[2], nWt[3], nbp[3], sums);

  // 6) mean-pool + linear head
  k_final<<<dim3(1), dim3(NG * NPRED), 0, stream>>>(sums, counts, linW, linb, out);
}
